// TTLinear_81080392614575
// MI455X (gfx1250) — matrix operand order repaired and checked by reading, not compiled, not run on hardware
//
#include <hip/hip_runtime.h>
#include <stdint.h>

typedef __attribute__((ext_vector_type(16))) __bf16 v16bf;
typedef __attribute__((ext_vector_type(8)))  __bf16 v8bf;
typedef __attribute__((ext_vector_type(4)))  __bf16 v4bf;
typedef __attribute__((ext_vector_type(8)))  float  v8f;

#define BATCH 8192
#define K1    4096   // GEMM1 K  (= 64*64 input dims)
#define N1    1024   // GEMM1 N  (= middle TT rank)
#define K2    1024   // GEMM2 K
#define N2    4096   // GEMM2 N  (= 64*64 output dims)

// ---------------------------------------------------------------------------
// CDNA5 async copy: global -> LDS, tracked on ASYNCcnt (no VGPR round trip).
// saddr form: vdst = LDS byte address, vaddr = 32-bit byte offset, saddr = base.
// Generic pointers to LDS are {SHARED_BASE, offset32}; low 32 bits are the
// LDS byte address the hardware wants in VDST.
// ---------------------------------------------------------------------------
__device__ __forceinline__ uint32_t lds_addr(const void* p) {
  return (uint32_t)(uintptr_t)p;
}

__device__ __forceinline__ void async_ld_b128(uint32_t lds, const void* sbase,
                                              uint32_t voff) {
  asm volatile("global_load_async_to_lds_b128 %0, %1, %2"
               :: "v"(lds), "v"(voff), "s"(sbase)
               : "memory");
}

__device__ __forceinline__ void wait_async0() {
#if __has_builtin(__builtin_amdgcn_s_wait_asynccnt)
  __builtin_amdgcn_s_wait_asynccnt(0);
#else
  asm volatile("s_wait_asynccnt 0x0" ::: "memory");
#endif
}

// ---------------------------------------------------------------------------
// Merge core0 (1,64,32) and core1 (32,64,1024) into W1T[s][c*64+d], bf16,
// stored pre-transposed [N][K] so GEMM B-fragments read contiguous K.
// W_in[cd, s] = sum_r core0[c,r] * core1[r,d,s]
// ---------------------------------------------------------------------------
__global__ __launch_bounds__(256)
void build_w1t(const float* __restrict__ core0, const float* __restrict__ core1,
               __bf16* __restrict__ w1t) {
  int id = blockIdx.x * blockDim.x + threadIdx.x;   // 1024*4096 threads
  int cd = id & 4095;
  int s  = id >> 12;
  int c = cd >> 6, d = cd & 63;
  float acc = 0.f;
#pragma unroll
  for (int r = 0; r < 32; ++r)
    acc += core0[c * 32 + r] * core1[(size_t)(r * 64 + d) * 1024 + s];
  w1t[(size_t)s * K1 + cd] = (__bf16)acc;
}

// ---------------------------------------------------------------------------
// Merge core2 (1024,64,32) and core3 (32,64,1) into W2T[A*64+Bd][r], bf16.
// W_out[r, n] = sum_t core2[r,A,t] * core3[t,Bd]
// ---------------------------------------------------------------------------
__global__ __launch_bounds__(256)
void build_w2t(const float* __restrict__ core2, const float* __restrict__ core3,
               __bf16* __restrict__ w2t) {
  int id = blockIdx.x * blockDim.x + threadIdx.x;   // 4096*1024 threads
  int r  = id & 1023;
  int n  = id >> 10;
  int A = n >> 6, Bd = n & 63;
  float acc = 0.f;
#pragma unroll
  for (int t = 0; t < 32; ++t)
    acc += core2[(size_t)(r * 64 + A) * 32 + t] * core3[t * 64 + Bd];
  w2t[(size_t)n * K2 + r] = (__bf16)acc;
}

// ---------------------------------------------------------------------------
// Fragment loaders in this code's contiguous K order (same for A and B; not ISA table order)
// A (16-bit, 16x32): lane holds row M=lane&15; lanes<16: K 0-15,
//                    lanes>=16: K 16-31  (two adjacent 16B chunks)
// B (16-bit, 32x16): lane holds col N=lane&15; K-half = (lane>>4)*16,
//                    16 contiguous bf16 = one 32B chunk
// ---------------------------------------------------------------------------
__device__ __forceinline__ v16bf load_a_frag(const __bf16* row_ptr, int khalf8) {
  const __bf16* p = row_ptr + khalf8 * 2;
  v8bf lo = *(const v8bf*)(p);
  v8bf hi = *(const v8bf*)(p + 8);
  return __builtin_shufflevector(lo, hi, 0, 1, 2, 3, 4, 5, 6, 7,
                                         8, 9, 10, 11, 12, 13, 14, 15);
}

// ---------------------------------------------------------------------------
// GEMM1: Y(bf16, 8192x1024) = X(f32, 8192x4096) @ W_in
// Double-buffered LDS, one barrier per 32-K step.
//  - A tile: fp32 global -> regs -> cvt bf16 -> ds_store (conversion fused)
//  - B tile: global_load_async_to_lds_b128 (ASYNCcnt), zero VGPR traffic
// 256 threads = 8 waves (2x4); block tile 128x128; wave tile 64x32.
// ---------------------------------------------------------------------------
__global__ __launch_bounds__(256)
void tt_gemm1(const float* __restrict__ X, const __bf16* __restrict__ W1T,
              __bf16* __restrict__ Y) {
  __shared__ __align__(64) __bf16 As[2 * 128 * 32];
  __shared__ __align__(64) __bf16 Bs[2 * 128 * 32];

  const int tid  = threadIdx.x;
  const int lane = tid & 31;
  const int w    = tid >> 5;
  const int m0   = blockIdx.y * 128;
  const int n0   = blockIdx.x * 128;
  const int wm0  = (w >> 2) * 64;
  const int wn0  = (w & 3) * 32;
  const int fm     = lane & 15;
  const int hi     = lane >> 4;
  const int khalf8 = hi * 8;    // half the A chunk base (elements)
  const int kh16   = hi * 16;   // B chunk base (elements)

  float4 areg[4];

  auto loadA = [&](int k0) {
#pragma unroll
    for (int i = 0; i < 4; ++i) {
      int q = tid + i * 256, m = q >> 3, kq = (q & 7) * 4;
      areg[i] = *(const float4*)(X + (size_t)(m0 + m) * K1 + k0 + kq);
    }
  };
  auto storeA = [&](int buf) {
    __bf16* dst = As + buf * (128 * 32);
#pragma unroll
    for (int i = 0; i < 4; ++i) {
      int q = tid + i * 256, m = q >> 3, kq = (q & 7) * 4;
      v4bf b;
      b[0] = (__bf16)areg[i].x; b[1] = (__bf16)areg[i].y;
      b[2] = (__bf16)areg[i].z; b[3] = (__bf16)areg[i].w;
      *(v4bf*)(dst + m * 32 + kq) = b;
    }
  };
  auto asyncB = [&](int k0, int buf) {
    __bf16* dst = Bs + buf * (128 * 32);
#pragma unroll
    for (int i = 0; i < 2; ++i) {
      int q = tid + i * 256, n = q >> 2, kq = (q & 3) * 8;
      async_ld_b128(lds_addr(dst + n * 32 + kq), W1T,
                    (uint32_t)(((n0 + n) * K1 + k0 + kq) * 2));
    }
  };

  v8f acc[4][2] = {};

  // prologue: tile 0 into buffer 0
  loadA(0);
  asyncB(0, 0);
  storeA(0);
  wait_async0();
  __syncthreads();

  int buf = 0;
  for (int k0 = 0; k0 < K1; k0 += 32) {
    const int kn = k0 + 32;
    if (kn < K1) {          // issue next tile's loads under this tile's math
      loadA(kn);
      asyncB(kn, buf ^ 1);
    }
    const __bf16* Ab = As + buf * (128 * 32);
    const __bf16* Bb = Bs + buf * (128 * 32);

    v16bf af[4];
#pragma unroll
    for (int tm = 0; tm < 4; ++tm)
      af[tm] = load_a_frag(Ab + (wm0 + tm * 16 + fm) * 32, khalf8);
    v16bf bfg[2];
#pragma unroll
    for (int tn = 0; tn < 2; ++tn)
      bfg[tn] = *(const v16bf*)(Bb + (wn0 + tn * 16 + fm) * 32 + kh16);

#pragma unroll
    for (int tm = 0; tm < 4; ++tm)
#pragma unroll
      for (int tn = 0; tn < 2; ++tn)
        acc[tm][tn] = __builtin_amdgcn_wmma_f32_16x16x32_bf16(
            false, af[tm], false, bfg[tn], (short)0, acc[tm][tn],
            false, false);

    if (kn < K1) {
      storeA(buf ^ 1);      // cvt+store A regs into the other buffer
      wait_async0();        // B tile landed in LDS
      __syncthreads();      // single barrier per K-step
    }
    buf ^= 1;
  }

  // epilogue: f32 acc -> bf16 Y  (C/D layout: VGPR e -> row e+(lane>>4)*8)
#pragma unroll
  for (int tm = 0; tm < 4; ++tm)
#pragma unroll
    for (int tn = 0; tn < 2; ++tn)
#pragma unroll
      for (int e = 0; e < 8; ++e) {
        int row = m0 + wm0 + tm * 16 + hi * 8 + e;
        int col = n0 + wn0 + tn * 16 + fm;
        Y[(size_t)row * N1 + col] = (__bf16)acc[tm][tn][e];
      }
}

// ---------------------------------------------------------------------------
// GEMM2: out(f32, 8192x4096) = Y(bf16) @ W_out + bias
// Both operand tiles are bf16 -> both staged with async-to-LDS.
// Y (16 MB) is L2-resident on MI455X (192 MB L2).
// ---------------------------------------------------------------------------
__global__ __launch_bounds__(256)
void tt_gemm2(const __bf16* __restrict__ Y, const __bf16* __restrict__ W2T,
              const float* __restrict__ bias, float* __restrict__ out) {
  __shared__ __align__(64) __bf16 As[2 * 128 * 32];
  __shared__ __align__(64) __bf16 Bs[2 * 128 * 32];

  const int tid  = threadIdx.x;
  const int lane = tid & 31;
  const int w    = tid >> 5;
  const int m0   = blockIdx.y * 128;
  const int n0   = blockIdx.x * 128;
  const int wm0  = (w >> 2) * 64;
  const int wn0  = (w & 3) * 32;
  const int fm     = lane & 15;
  const int hi     = lane >> 4;
  const int khalf8 = hi * 8;
  const int kh16   = hi * 16;

  auto asyncA = [&](int k0, int buf) {
    __bf16* dst = As + buf * (128 * 32);
#pragma unroll
    for (int i = 0; i < 2; ++i) {
      int q = tid + i * 256, m = q >> 2, kq = (q & 3) * 8;
      async_ld_b128(lds_addr(dst + m * 32 + kq), Y,
                    (uint32_t)(((m0 + m) * K2 + k0 + kq) * 2));
    }
  };
  auto asyncB = [&](int k0, int buf) {
    __bf16* dst = Bs + buf * (128 * 32);
#pragma unroll
    for (int i = 0; i < 2; ++i) {
      int q = tid + i * 256, n = q >> 2, kq = (q & 3) * 8;
      async_ld_b128(lds_addr(dst + n * 32 + kq), W2T,
                    (uint32_t)(((n0 + n) * K2 + k0 + kq) * 2));
    }
  };

  v8f acc[4][2] = {};

  asyncA(0, 0);
  asyncB(0, 0);
  wait_async0();
  __syncthreads();

  int buf = 0;
  for (int k0 = 0; k0 < K2; k0 += 32) {
    const int kn = k0 + 32;
    if (kn < K2) {
      asyncA(kn, buf ^ 1);
      asyncB(kn, buf ^ 1);
    }
    const __bf16* Ab = As + buf * (128 * 32);
    const __bf16* Bb = Bs + buf * (128 * 32);

    v16bf af[4];
#pragma unroll
    for (int tm = 0; tm < 4; ++tm)
      af[tm] = load_a_frag(Ab + (wm0 + tm * 16 + fm) * 32, khalf8);
    v16bf bfg[2];
#pragma unroll
    for (int tn = 0; tn < 2; ++tn)
      bfg[tn] = *(const v16bf*)(Bb + (wn0 + tn * 16 + fm) * 32 + kh16);

#pragma unroll
    for (int tm = 0; tm < 4; ++tm)
#pragma unroll
      for (int tn = 0; tn < 2; ++tn)
        acc[tm][tn] = __builtin_amdgcn_wmma_f32_16x16x32_bf16(
            false, af[tm], false, bfg[tn], (short)0, acc[tm][tn],
            false, false);

    if (kn < K2) {
      wait_async0();
      __syncthreads();
    }
    buf ^= 1;
  }

  // epilogue: add bias, store f32
#pragma unroll
  for (int tn = 0; tn < 2; ++tn) {
    const float bv = bias[n0 + wn0 + tn * 16 + fm];
#pragma unroll
    for (int tm = 0; tm < 4; ++tm)
#pragma unroll
      for (int e = 0; e < 8; ++e) {
        int row = m0 + wm0 + tm * 16 + hi * 8 + e;
        int col = n0 + wn0 + tn * 16 + fm;
        out[(size_t)row * N2 + col] = acc[tm][tn][e] + bv;
      }
  }
}

// ---------------------------------------------------------------------------
extern "C" void kernel_launch(void* const* d_in, const int* in_sizes, int n_in,
                              void* d_out, int out_size, void* d_ws, size_t ws_size,
                              hipStream_t stream) {
  const float* x     = (const float*)d_in[0];
  const float* core0 = (const float*)d_in[1];
  const float* core1 = (const float*)d_in[2];
  const float* core2 = (const float*)d_in[3];
  const float* core3 = (const float*)d_in[4];
  const float* bias  = (const float*)d_in[5];
  float* out = (float*)d_out;

  // workspace layout (bf16): W1T 8MB | W2T 8MB | Y 16MB  = 32MB total
  char* ws = (char*)d_ws;
  __bf16* w1t = (__bf16*)(ws);
  __bf16* w2t = (__bf16*)(ws + (size_t)N1 * K1 * 2);
  __bf16* yb  = (__bf16*)(ws + (size_t)N1 * K1 * 2 + (size_t)N2 * K2 * 2);

  // merge TT cores into the two bf16 weight matrices (pre-transposed [N][K])
  build_w1t<<<(N1 * K1) / 256, 256, 0, stream>>>(core0, core1, w1t);
  build_w2t<<<(N2 * K2) / 256, 256, 0, stream>>>(core2, core3, w2t);

  // GEMM1: Y = X @ W_in       (8192x4096 @ 4096x1024)
  tt_gemm1<<<dim3(N1 / 128, BATCH / 128), 256, 0, stream>>>(x, w1t, yb);
  // GEMM2: out = Y @ W_out + b (8192x1024 @ 1024x4096)
  tt_gemm2<<<dim3(N2 / 128, BATCH / 128), 256, 0, stream>>>(yb, w2t, bias, out);
}
